// EncoderNetwork_12721693131090
// MI455X (gfx1250) — matrix operand order repaired and checked by reading, not compiled, not run on hardware
//
#include <hip/hip_runtime.h>

// CDNA5 / gfx1250 DAG-GNN encoder, transposed split-K formulation.
// Dt = Wt x Xt per layer via v_wmma_f32_16x16x32_f16. The K<->feature map of
// the B operand is chosen so that the WMMA D layout (lane=node, VGPR=feature)
// IS the next layer's B layout: features 0-7 live at K=0-7 (half-0 lanes),
// features 8-15 at K=16-23 (half-1 lanes). Transposed weights + biases are
// pre-packed once (A fragments use the same map; bias folded into C operand).
// Layer transitions are relu + 4 packed f16 converts: no shuffles, no LDS.

typedef __attribute__((ext_vector_type(16))) _Float16 v16h;
typedef __attribute__((ext_vector_type(2)))  __fp16   v2fp16;
typedef __attribute__((ext_vector_type(8)))  float    v8f;

#define WPB 8             // waves per block
#define TPB (WPB * 32)    // 256 threads = 8 wave32
#define SLOT 2048         // bytes per packed layer slot: 1KB A-frag + 1KB C-init

__device__ __forceinline__ v8f wmma_f16(v16h a, v16h b, v8f c) {
  return __builtin_amdgcn_wmma_f32_16x16x32_f16(false, a, false, b, (short)0, c,
                                                false, false);
}

struct MlpW { const float *w1, *b1, *w2, *b2, *w3, *b3; };

__device__ __forceinline__ int pack2(float a, float b) {
#if __has_builtin(__builtin_amdgcn_cvt_pkrtz)
  union { v2fp16 h; int i; } u;
  u.h = __builtin_amdgcn_cvt_pkrtz(a, b);
  return u.i;
#else
  union { _Float16 h[2]; int i; } u;
  u.h[0] = (_Float16)a; u.h[1] = (_Float16)b;
  return u.i;
#endif
}

__device__ __forceinline__ float relu(float x) { return x > 0.0f ? x : 0.0f; }

// Single-instruction lane swizzles (group-of-32 XOR patterns).
template <int IMM>
__device__ __forceinline__ float swzf(float v) {
  union { float f; int i; } u;
  u.f = v;
  u.i = __builtin_amdgcn_ds_swizzle(u.i, IMM);
  return u.f;
}

// Split-K map: K -> feature. Standard layers: [0,8)->k, [16,24)->k-8.
// The kin=21 layer additionally maps [8,13) -> k+8 (features 16..20).
__device__ __forceinline__ int kmap(int k, int kin) {
  if (k < 8) return k;
  if (k >= 16 && k < 24) return k - 8;
  if (kin == 21 && k >= 8 && k < 13) return k + 8;
  return -1;
}

// ---------------- one-time packing of Wt A-fragments + bias C-init ----------
// A layout (16-bit A 16x32): lane m = row (output feature), elem e ->
// K = e + 16*half.  A[m][k] = W[kmap(k)*nout + m].
// C/D layout: lane n, VGPR r -> (M = r + 8*half, N = n): C-init[r] = bias[r+8*half].
__global__ void k_pack(MlpW p, MlpW m, MlpW u, MlpW d, MlpW g,
                       char* __restrict__ out) {
  const int w = blockIdx.x;       // 0..14 : (mlp, layer)
  const int lane = threadIdx.x;   // 32
  const int mlp = w / 3, layer = w % 3;
  MlpW W = (mlp == 0) ? p : (mlp == 1) ? m : (mlp == 2) ? u : (mlp == 3) ? d : g;
  const float* wp; const float* bp;
  int kin, nout;
  if (layer == 0) { wp = W.w1; bp = W.b1; nout = 16; kin = (mlp == 0) ? 5 : (mlp == 3) ? 21 : 16; }
  else if (layer == 1) { wp = W.w2; bp = W.b2; kin = 16; nout = 8; }
  else { wp = W.w3; bp = W.b3; kin = 8; nout = 16; }
  const int mrow = lane & 15, half = lane >> 4;
  _Float16* af = (_Float16*)(out + (long long)w * SLOT) + lane * 16;
  for (int e = 0; e < 16; ++e) {
    int k = e + 16 * half;
    int f = kmap(k, kin);
    float v = (f >= 0 && f < kin && mrow < nout) ? wp[f * nout + mrow] : 0.0f;
    af[e] = (_Float16)v;
  }
  float* ci = (float*)(out + (long long)w * SLOT + 1024) + lane * 8;
  for (int r = 0; r < 8; ++r) {
    int f = r + 8 * half;
    ci[r] = (f < nout) ? bp[f] : 0.0f;
  }
}

__device__ __forceinline__ v16h load_afrag(const char* __restrict__ wb, int slot,
                                           int lane) {
  return *(const v16h*)(wb + (long long)slot * SLOT + lane * 32);
}
__device__ __forceinline__ v8f load_cinit(const char* __restrict__ wb, int slot,
                                          int lane) {
  return *(const v8f*)(wb + (long long)slot * SLOT + 1024 + lane * 32);
}

__device__ __forceinline__ v16h bfrag_from_ints(const int* pi) {
  union { int i[8]; v16h h; } b;
#pragma unroll
  for (int j = 0; j < 8; ++j) b.i[j] = pi[j];
  return b.h;
}

// ---------------- activation B-fragment builders (branchless) ---------------
// Split-K B layout: lane (n, half) carries features 8*half+0..7 in elems 0..7.

// Kin=16, stride 16 f32 rows: 2 x float4 per lane, no masking.
__device__ __forceinline__ v16h load_B16(const float* __restrict__ act,
                                         long long row0, int nval, int lane) {
  const int n = lane & 15, half = lane >> 4;
  const int nc = (n < nval) ? n : (nval - 1);          // clamp: dup cols harmless
  const float4* r = (const float4*)(act + (row0 + nc) * 16ll + 8 * half);
  float4 u0 = r[0], u1 = r[1];
  int pi[8] = {};
  pi[0] = pack2(u0.x, u0.y); pi[1] = pack2(u0.z, u0.w);
  pi[2] = pack2(u1.x, u1.y); pi[3] = pack2(u1.z, u1.w);
  return bfrag_from_ints(pi);
}

// Kin=5, stride 5 f32 rows (prep): features 0..4 -> half-0 lanes only.
__device__ __forceinline__ v16h load_B5(const float* __restrict__ act,
                                        long long row0, int nval, int lane) {
  const int n = lane & 15, half = lane >> 4;
  const int nc = (n < nval) ? n : (nval - 1);
  const float* r = act + (row0 + nc) * 5ll;
  float v0 = r[0], v1 = r[1], v2 = r[2], v3 = r[3], v4 = r[4];
  int pi[8] = {};
  pi[0] = half ? 0 : pack2(v0, v1);
  pi[1] = half ? 0 : pack2(v2, v3);
  pi[2] = half ? 0 : pack2(v4, 0.0f);
  return bfrag_from_ints(pi);
}

// Kin=21: concat(x[:,0:5], h[:,0:16]). Features e+8*half in elems 0..7;
// features 16..20 in elems 8..12 of half-0 lanes (K=8..12).
__device__ __forceinline__ v16h load_B21(const float* __restrict__ x,
                                         const float* __restrict__ h,
                                         long long row0, int nval, int lane) {
  const int n = lane & 15, half = lane >> 4;
  const int nc = (n < nval) ? n : (nval - 1);
  const float* xr = x + (row0 + nc) * 5ll;
  const float* hr = h + (row0 + nc) * 16ll;
  float v[13];
#pragma unroll
  for (int e = 0; e < 8; ++e) {
    int f = e + 8 * half;
    const float* p = (f < 5) ? (xr + f) : (hr + (f - 5));
    v[e] = *p;
  }
#pragma unroll
  for (int e = 8; e < 13; ++e) v[e] = half ? 0.0f : hr[e + 3];  // features 16..20
  int pi[8] = {};
  pi[0] = pack2(v[0], v[1]);  pi[1] = pack2(v[2], v[3]);
  pi[2] = pack2(v[4], v[5]);  pi[3] = pack2(v[6], v[7]);
  pi[4] = pack2(v[8], v[9]);  pi[5] = pack2(v[10], v[11]);
  pi[6] = pack2(v[12], 0.0f);
  return bfrag_from_ints(pi);
}

// Layer transition: D layout == split-K B layout. relu + 4 packed converts.
__device__ __forceinline__ v16h transB(v8f d) {
  int pi[8] = {};
  pi[0] = pack2(relu(d[0]), relu(d[1]));
  pi[1] = pack2(relu(d[2]), relu(d[3]));
  pi[2] = pack2(relu(d[4]), relu(d[5]));
  pi[3] = pack2(relu(d[6]), relu(d[7]));
  return bfrag_from_ints(pi);
}

// 3-layer MLP on one 16-node tile; biases folded into C-init; out = Dt3.
__device__ __forceinline__ v8f mlp3(v16h bact, const char* __restrict__ wb,
                                    int slot0, int lane) {
  v16h a1 = load_afrag(wb, slot0 + 0, lane);
  v16h a2 = load_afrag(wb, slot0 + 1, lane);
  v16h a3 = load_afrag(wb, slot0 + 2, lane);
  v8f c1 = load_cinit(wb, slot0 + 0, lane);
  v8f c2 = load_cinit(wb, slot0 + 1, lane);
  v8f c3 = load_cinit(wb, slot0 + 2, lane);
  v8f d1 = wmma_f16(a1, bact, c1);
  v8f d2 = wmma_f16(a2, transB(d1), c2);
  v8f d3 = wmma_f16(a3, transB(d2), c3);
  return d3;
}

// ---------------- kernels ---------------------------------------------------

__global__ void k_zero(float* __restrict__ p, long long n) {
  long long i = (long long)blockIdx.x * blockDim.x + threadIdx.x;
  if (i < n) p[i] = 0.0f;
}

// out[rows,16] = MLP(in[rows,:KIN]); lane owns node row0+(lane&15),
// registers own 8 contiguous features -> two float4 stores.
template <int KIN>
__global__ void k_node_mlp(const float* __restrict__ in, float* __restrict__ out,
                           const char* __restrict__ wb, int slot0, long long nrows) {
  const int lane = threadIdx.x & 31;
  const int wv = threadIdx.x >> 5;
  long long row0 = ((long long)blockIdx.x * WPB + wv) * 16;
  if (row0 >= nrows) return;                 // wave-uniform; EXEC all-1s for WMMA
  const int nval = (int)((nrows - row0 < 16) ? (nrows - row0) : 16);

  v16h b = (KIN == 5) ? load_B5(in, row0, nval, lane)
                      : load_B16(in, row0, nval, lane);
  v8f o = mlp3(b, wb, slot0, lane);

  const int n = lane & 15, half = lane >> 4;
  long long node = row0 + n;
  if (node < nrows) {
    float4* op = (float4*)(out + node * 16 + 8 * half);
    op[0] = make_float4(o[0], o[1], o[2], o[3]);
    op[1] = make_float4(o[4], o[5], o[6], o[7]);
  }
}

// One thread per edge: agg[dst] += y[src] * mask; deg[dst] += mask.
__global__ void k_scatter(const int* __restrict__ dst, const int* __restrict__ src,
                          const int* __restrict__ mask, const float* __restrict__ y,
                          float* __restrict__ agg, float* __restrict__ deg,
                          long long E) {
  long long e = (long long)blockIdx.x * blockDim.x + threadIdx.x;
  if (e >= E) return;
  if (mask[e] == 0) return;
  long long d = dst[e], s = src[e];
  const float4* ys = (const float4*)(y + s * 16);
  float* ad = agg + d * 16;
#pragma unroll
  for (int i = 0; i < 4; ++i) {
    float4 v = ys[i];
    atomicAdd(ad + 4 * i + 0, v.x);
    atomicAdd(ad + 4 * i + 1, v.y);
    atomicAdd(ad + 4 * i + 2, v.z);
    atomicAdd(ad + 4 * i + 3, v.w);
  }
  atomicAdd(deg + d, 1.0f);
}

// h += (deg[node]>0) * MLP_u(agg), in place; node mask is per-lane uniform.
__global__ void k_update(const float* __restrict__ agg, const float* __restrict__ deg,
                         float* __restrict__ h, const char* __restrict__ wb,
                         int slot0, long long nrows) {
  const int lane = threadIdx.x & 31;
  const int wv = threadIdx.x >> 5;
  long long row0 = ((long long)blockIdx.x * WPB + wv) * 16;
  if (row0 >= nrows) return;
  const int nval = (int)((nrows - row0 < 16) ? (nrows - row0) : 16);

  v16h b = load_B16(agg, row0, nval, lane);
  v8f o = mlp3(b, wb, slot0, lane);

  const int n = lane & 15, half = lane >> 4;
  long long node = row0 + n;
  long long nodec = (node < nrows) ? node : (nrows - 1);
  float nm = (deg[nodec] > 0.0f) ? 1.0f : 0.0f;
  if (node < nrows) {
    float4* hp = (float4*)(h + node * 16 + 8 * half);
    float4 a0 = hp[0], a1 = hp[1];
    hp[0] = make_float4(a0.x + nm * o[0], a0.y + nm * o[1],
                        a0.z + nm * o[2], a0.w + nm * o[3]);
    hp[1] = make_float4(a1.x + nm * o[4], a1.y + nm * o[5],
                        a1.z + nm * o[6], a1.w + nm * o[7]);
  }
}

// z = MLP_d(concat(x, h))
__global__ void k_dag_mlp(const float* __restrict__ x, const float* __restrict__ h,
                          float* __restrict__ z, const char* __restrict__ wb,
                          int slot0, long long nrows) {
  const int lane = threadIdx.x & 31;
  const int wv = threadIdx.x >> 5;
  long long row0 = ((long long)blockIdx.x * WPB + wv) * 16;
  if (row0 >= nrows) return;
  const int nval = (int)((nrows - row0 < 16) ? (nrows - row0) : 16);

  v16h b = load_B21(x, h, row0, nval, lane);
  v8f o = mlp3(b, wb, slot0, lane);

  const int n = lane & 15, half = lane >> 4;
  long long node = row0 + n;
  if (node < nrows) {
    float4* zp = (float4*)(z + node * 16 + 8 * half);
    zp[0] = make_float4(o[0], o[1], o[2], o[3]);
    zp[1] = make_float4(o[4], o[5], o[6], o[7]);
  }
}

// One wave per dag: dag[j,:] = sum_{i in [ptr[j],ptr[j+1])} z[i,:]
__global__ void k_segsum(const float* __restrict__ z, const int* __restrict__ ptr,
                         float* __restrict__ dag, int ndags) {
  int j = blockIdx.x;
  if (j >= ndags) return;
  int lane = threadIdx.x;
  int beg = ptr[j], end = ptr[j + 1];
  int dim = lane & 15, off = lane >> 4;
  float s = 0.0f;
  for (int i = beg + off; i < end; i += 2) s += z[(long long)i * 16 + dim];
  s += swzf<0x401F>(s);                       // SWAPX16: combine lane halves
  if (lane < 16) dag[(long long)j * 16 + dim] = s;
}

// gsum[0:16] += column sums of MLP_g(dag): ds_swizzle butterfly over the 16
// node lanes, then one atomic per feature per wave.
__global__ void k_glob_mlp(const float* __restrict__ dag, float* __restrict__ gsum,
                           const char* __restrict__ wb, int slot0, long long nrows) {
  const int lane = threadIdx.x & 31;
  const int wv = threadIdx.x >> 5;
  long long row0 = ((long long)blockIdx.x * WPB + wv) * 16;
  if (row0 >= nrows) return;
  const int nval = (int)((nrows - row0 < 16) ? (nrows - row0) : 16);

  v16h b = load_B16(dag, row0, nval, lane);
  v8f o = mlp3(b, wb, slot0, lane);

  const int n = lane & 15, half = lane >> 4;
  long long node = row0 + n;
  float v[8];
#pragma unroll
  for (int r = 0; r < 8; ++r) v[r] = (node < nrows) ? o[r] : 0.0f;
#pragma unroll
  for (int r = 0; r < 8; ++r) {
    v[r] += swzf<0x041F>(v[r]);               // xor 1
    v[r] += swzf<0x081F>(v[r]);               // xor 2
    v[r] += swzf<0x101F>(v[r]);               // xor 4
    v[r] += swzf<0x201F>(v[r]);               // xor 8
  }
  if (n == 0) {
#pragma unroll
    for (int r = 0; r < 8; ++r) atomicAdd(&gsum[r + 8 * half], v[r]);
  }
}

// ---------------- launch ----------------------------------------------------

extern "C" void kernel_launch(void* const* d_in, const int* in_sizes, int n_in,
                              void* d_out, int out_size, void* d_ws, size_t ws_size,
                              hipStream_t stream) {
  (void)n_in; (void)out_size; (void)ws_size;

  const float* x          = (const float*)d_in[0];
  const int*   edge_index = (const int*)d_in[1];
  const int*   mask       = (const int*)d_in[2];   // bool mask widened to int
  const int*   ptr        = (const int*)d_in[3];

  const long long N = in_sizes[0] / 5;
  const long long E = in_sizes[1] / 2;
  const int Drounds = (int)(in_sizes[2] / E);
  const int ndags   = in_sizes[3] - 1;

  auto getW = [&](int b) {
    return MlpW{(const float*)d_in[b],     (const float*)d_in[b + 1],
                (const float*)d_in[b + 2], (const float*)d_in[b + 3],
                (const float*)d_in[b + 4], (const float*)d_in[b + 5]};
  };
  MlpW Wp = getW(4), Wm = getW(10), Wu = getW(16), Wd = getW(22), Wg = getW(28);

  float* out  = (float*)d_out;
  float* h    = out;                                   // node_embeddings [N,16]
  float* dag  = out + N * 16;                          // dag_summaries [ndags,16]
  float* gsum = out + N * 16 + (long long)ndags * 16;  // global [1,16]

  // ws layout: packed slots (15 * 2KB = 30KB, round to 32KB), then y [N,16],
  // agg [N,16], deg [N] (agg/deg contiguous for one zero pass).
  char*  wb  = (char*)d_ws;
  float* y   = (float*)((char*)d_ws + 32768);
  float* agg = y + N * 16;
  float* deg = agg + N * 16;

  const int* dst = edge_index;
  const int* src = edge_index + E;

  const long long tiles = (N + 15) / 16;
  const int mlpGrid = (int)((tiles + WPB - 1) / WPB);
  const long long zn = N * 16 + N;
  const int zGrid = (int)((zn + 255) / 256);
  const int sGrid = (int)((E + 255) / 256);

  // Pack Wt A-fragments + bias C-init images (once per launch).
  k_pack<<<15, 32, 0, stream>>>(Wp, Wm, Wu, Wd, Wg, wb);

  // NodeEncoder prep: h = MLP_p(x)     (slots: p=0, m=3, u=6, d=9, g=12)
  k_node_mlp<5><<<mlpGrid, TPB, 0, stream>>>(x, h, wb, 0, N);

  // D rounds of masked message passing
  for (int r = 0; r < Drounds; ++r) {
    k_node_mlp<16><<<mlpGrid, TPB, 0, stream>>>(h, y, wb, 3, N);
    k_zero<<<zGrid, 256, 0, stream>>>(agg, zn);
    k_scatter<<<sGrid, 256, 0, stream>>>(dst, src, mask + (long long)r * E, y, agg,
                                         deg, E);
    k_update<<<mlpGrid, TPB, 0, stream>>>(agg, deg, h, wb, 6, N);
  }

  // DagEncoder: z = MLP_d([x,h]) then contiguous segment sums
  k_dag_mlp<<<mlpGrid, TPB, 0, stream>>>(x, h, y /*reuse as z*/, wb, 9, N);
  k_segsum<<<ndags, 32, 0, stream>>>(y, ptr, dag, ndags);

  // GlobalEncoder: sum over all dags of MLP_g(dag)
  k_zero<<<1, 32, 0, stream>>>(gsum, 16);
  const long long dtiles = ((long long)ndags + 15) / 16;
  const int gGrid = (int)((dtiles + WPB - 1) / WPB);
  k_glob_mlp<<<gGrid, TPB, 0, stream>>>(dag, gsum, wb, 12, (long long)ndags);
}
